// ConcatCriticCNN_22986664968235
// MI455X (gfx1250) — matrix operand order repaired and checked by reading, not compiled, not run on hardware
//
#include <hip/hip_runtime.h>

// ---------------- CDNA5 WMMA types ----------------
typedef __attribute__((ext_vector_type(16))) __bf16 v16bf;
typedef __attribute__((ext_vector_type(8)))  float  v8f;

union FragBF {
    uint4 q[2];
    v16bf v;
};

__device__ __forceinline__ unsigned short f2bf(float f) {
    union { float f; unsigned int u; } c;
    c.f = f;
    unsigned int u = c.u;
    unsigned int rnd = 0x7FFFu + ((u >> 16) & 1u);   // round-to-nearest-even
    return (unsigned short)((u + rnd) >> 16);
}

// ---------------- tiny utility kernels ----------------
__global__ void zero_kernel(float* p, int n) {
    int i = blockIdx.x * blockDim.x + threadIdx.x;
    if (i < n) p[i] = 0.0f;
}

__global__ void cvt_bf16_kernel(const float* __restrict__ in, unsigned short* __restrict__ out, int n) {
    int i = blockIdx.x * blockDim.x + threadIdx.x;
    if (i < n) out[i] = f2bf(in[i]);
}

// xe[i,d] = bf[d] + (1/1024) * sum_c pool[i,c] * wf[d,c]
__global__ void fc_kernel(const float* __restrict__ pool, const float* __restrict__ wf,
                          const float* __restrict__ bf, float* __restrict__ e) {
    int idx = blockIdx.x * blockDim.x + threadIdx.x;
    if (idx >= 256 * 128) return;
    int ii = idx >> 7, d = idx & 127;
    float acc = 0.0f;
    const float* prow = pool + ii * 128;
    const float* wrow = wf + d * 128;
    for (int c = 0; c < 128; ++c) acc += prow[c] * wrow[c];
    e[idx] = bf[d] + acc * (1.0f / 1024.0f);
}

// part[i,h] = sum_d e[i,d] * w1[h, colOff + d]   (w1 rows are length 256)
__global__ void part_kernel(const float* __restrict__ e, const float* __restrict__ w1,
                            int colOff, float* __restrict__ outp) {
    int idx = blockIdx.x * blockDim.x + threadIdx.x;
    if (idx >= 256 * 512) return;
    int ii = idx >> 9, h = idx & 511;
    const float* erow = e + ii * 128;
    const float* wrow = w1 + h * 256 + colOff;
    float acc = 0.0f;
    for (int d = 0; d < 128; ++d) acc += erow[d] * wrow[d];
    outp[idx] = acc;
}

// ---------------- fused conv tower: conv1 -> relu -> conv2(WMMA) -> relu -> pool-sum ----------------
// grid: (16 t-tiles, 256 batches), block 256 (8 waves). Each block: 128(co) x 64(t) conv2 tile.
__global__ __launch_bounds__(256) void tower_kernel(
    const float* __restrict__ x,          // (256, 8, 1024)
    const float* __restrict__ w1,         // (64, 8, 5)
    const float* __restrict__ b1,         // (64)
    const unsigned short* __restrict__ w2h, // (128, 320) bf16 im2col weights
    const float* __restrict__ b2,         // (128)
    float* __restrict__ pool)             // (256, 128) pool sums (atomic accum)
{
    const int tb = blockIdx.x * 64;       // t base of output tile
    const int b  = blockIdx.y;
    const int tid  = threadIdx.x;
    const int lane = tid & 31;
    const int wv   = tid >> 5;            // 0..7

    __shared__ float sTemp[64][68];                         // conv1(relu) slice, t = tb-2 .. tb+65
    __shared__ __align__(16) unsigned short sB[64][320];    // im2col tile [t_local][ci*5+dk], bf16

    // Phase 1: conv1 + relu into sTemp  (64*68 = 4352 values, 17 per thread)
    for (int i = tid; i < 64 * 68; i += 256) {
        int co1 = i / 68;
        int ul  = i % 68;
        int t   = tb - 2 + ul;            // conv1 output time index
        float v = 0.0f;
        if (t >= 0 && t < 1024) {
            float acc = b1[co1];
            const float* wr = w1 + co1 * 40;
            const float* xb = x + b * 8192;
            for (int ci = 0; ci < 8; ++ci) {
                const float* xr = xb + ci * 1024;
                for (int dk = 0; dk < 5; ++dk) {
                    int u = t + dk - 2;
                    if (u >= 0 && u < 1024) acc += xr[u] * wr[ci * 5 + dk];
                }
            }
            v = fmaxf(acc, 0.0f);
        }
        sTemp[co1][ul] = v;
    }
    __syncthreads();

    // Phase 2: build im2col B tile (bf16): sB[tl][ci*5+dk] = sTemp[ci][tl+dk]
    for (int i = tid; i < 64 * 320; i += 256) {
        int tl   = i / 320;
        int kidx = i % 320;
        int ci   = kidx / 5;
        int dk   = kidx % 5;
        sB[tl][kidx] = f2bf(sTemp[ci][tl + dk]);
    }
    __syncthreads();

    // Phase 3: WMMA GEMM 128(co) x 64(t) x 320(k); wave wv owns co rows [16*wv, 16*wv+16)
    v8f c0 = {}, c1 = {}, c2 = {}, c3 = {};
    const int mrow   = wv * 16 + (lane & 15);       // A-matrix row (co)
    const int baseK8 = (lane >> 4) * 16;            // A frag second-half-wave K offset
    const int bK16   = (lane >> 4) * 16;            // B frag second-half-wave K offset

    for (int kb = 0; kb < 320; kb += 32) {
        FragBF fa;
        const uint4* pa = reinterpret_cast<const uint4*>(w2h + mrow * 320 + kb + baseK8);
        fa.q[0] = pa[0];
        fa.q[1] = *reinterpret_cast<const uint4*>(w2h + mrow * 320 + kb + baseK8 + 8);

        #pragma unroll
        for (int n = 0; n < 4; ++n) {
            FragBF fb;
            const uint4* pb = reinterpret_cast<const uint4*>(&sB[n * 16 + (lane & 15)][kb + bK16]);
            fb.q[0] = pb[0];
            fb.q[1] = pb[1];
            v8f* cp = (n == 0) ? &c0 : (n == 1) ? &c1 : (n == 2) ? &c2 : &c3;
            *cp = __builtin_amdgcn_wmma_f32_16x16x32_bf16(
                false, fa.v, false, fb.v, (short)0, *cp, false, false);
        }
    }

    // Phase 4: bias + relu + partial pool sums, reduce across 16 lanes, atomic accumulate
    #pragma unroll
    for (int r = 0; r < 8; ++r) {
        int co = wv * 16 + r + ((lane >> 4) << 3);
        float bias = b2[co];
        float s = fmaxf(c0[r] + bias, 0.0f) + fmaxf(c1[r] + bias, 0.0f)
                + fmaxf(c2[r] + bias, 0.0f) + fmaxf(c3[r] + bias, 0.0f);
        // reduce over the 16 lanes that hold the 16 t-columns of this row group
        for (int off = 8; off >= 1; off >>= 1) s += __shfl_xor(s, off, 16);
        if ((lane & 15) == 0) atomicAdd(&pool[b * 128 + co], s);
    }
}

// ---------------- pair kernel: h1(on the fly) -> WMMA vs h_w2 -> relu -> dot(w3) ----------------
// grid: 1024 blocks x 128 threads (4 waves). Block = 64 consecutive (i,j) rows, i constant.
__global__ __launch_bounds__(128) void pair_kernel(
    const float* __restrict__ xp,   // (256, 512)
    const float* __restrict__ yp,   // (256, 512)
    const float* __restrict__ b1,   // h_b1 (512)
    const unsigned short* __restrict__ W2h, // h_w2 bf16 (512, 512)
    const float* __restrict__ b2,   // h_b2 (512)
    const float* __restrict__ w3,   // h_w3 (512)
    const float* __restrict__ b3,   // h_b3 (1)
    float* __restrict__ out)        // scores, 65536 rows
{
    const int rowbase = blockIdx.x * 64;
    const int iIdx    = rowbase >> 8;
    const int jbase   = rowbase & 255;
    const int tid  = threadIdx.x;
    const int lane = tid & 31;
    const int wv   = tid >> 5;      // 0..3

    __shared__ __align__(16) unsigned short sA[64][512];    // h1 tile, bf16 (64 KB)

    // Stage A tile: sA[rl][k] = bf16(relu(xp[i,k] + yp[jbase+rl,k] + b1[k]))
    const float* xrow = xp + iIdx * 512;
    for (int idx = tid; idx < 64 * 512; idx += 128) {
        int rl = idx >> 9;
        int k  = idx & 511;
        float v = xrow[k] + yp[(jbase + rl) * 512 + k] + b1[k];
        sA[rl][k] = f2bf(fmaxf(v, 0.0f));
    }
    __syncthreads();

    const int rlA    = wv * 16 + (lane & 15);   // A row (local)
    const int baseK8 = (lane >> 4) * 16;
    const int bK16   = (lane >> 4) * 16;
    const float b3v  = b3[0];

    float part[8];
    #pragma unroll
    for (int r = 0; r < 8; ++r) part[r] = 0.0f;

    for (int g = 0; g < 32; ++g) {
        const int gcol = g * 16 + (lane & 15);        // h_w2 row = h2 column
        const unsigned short* wrow = W2h + gcol * 512;
        if (g < 31)  // prefetch next panel of h_w2 (gfx1250 global_prefetch path)
            __builtin_prefetch(W2h + (gcol + 16) * 512, 0, 0);

        v8f c = {};
        #pragma unroll 4
        for (int kk = 0; kk < 16; ++kk) {
            const int kb = kk * 32;
            FragBF fa, fb;
            const unsigned short* ap = &sA[rlA][kb + baseK8];
            fa.q[0] = *reinterpret_cast<const uint4*>(ap);
            fa.q[1] = *reinterpret_cast<const uint4*>(ap + 8);
            const uint4* bp = reinterpret_cast<const uint4*>(wrow + kb + bK16);
            fb.q[0] = bp[0];
            fb.q[1] = bp[1];
            c = __builtin_amdgcn_wmma_f32_16x16x32_bf16(
                false, fa.v, false, fb.v, (short)0, c, false, false);
        }

        float bias = b2[gcol];
        float t3   = w3[gcol];
        #pragma unroll
        for (int r = 0; r < 8; ++r)
            part[r] += fmaxf(c[r] + bias, 0.0f) * t3;
    }

    // Reduce over the 16 lanes holding the 16 h2-column residues for each row group
    #pragma unroll
    for (int r = 0; r < 8; ++r) {
        float s = part[r];
        for (int off = 8; off >= 1; off >>= 1) s += __shfl_xor(s, off, 16);
        if ((lane & 15) == 0) {
            int row = rowbase + wv * 16 + r + ((lane >> 4) << 3);
            out[row] = s + b3v;
        }
    }
}

// ---------------- host side ----------------
extern "C" void kernel_launch(void* const* d_in, const int* in_sizes, int n_in,
                              void* d_out, int out_size, void* d_ws, size_t ws_size,
                              hipStream_t stream) {
    const float* x     = (const float*)d_in[0];
    const float* y     = (const float*)d_in[1];
    const float* cx_w1 = (const float*)d_in[2];
    const float* cx_b1 = (const float*)d_in[3];
    const float* cx_w2 = (const float*)d_in[4];
    const float* cx_b2 = (const float*)d_in[5];
    const float* cx_wf = (const float*)d_in[6];
    const float* cx_bf = (const float*)d_in[7];
    const float* cy_w1 = (const float*)d_in[8];
    const float* cy_b1 = (const float*)d_in[9];
    const float* cy_w2 = (const float*)d_in[10];
    const float* cy_b2 = (const float*)d_in[11];
    const float* cy_wf = (const float*)d_in[12];
    const float* cy_bf = (const float*)d_in[13];
    const float* h_w1  = (const float*)d_in[14];
    const float* h_b1  = (const float*)d_in[15];
    const float* h_w2  = (const float*)d_in[16];
    const float* h_b2  = (const float*)d_in[17];
    const float* h_w3  = (const float*)d_in[18];
    const float* h_b3  = (const float*)d_in[19];
    float* out = (float*)d_out;

    char* ws = (char*)d_ws;
    float* pool_x          = (float*)(ws + 0);          // 256*128 f32
    float* pool_y          = (float*)(ws + 131072);
    float* xe              = (float*)(ws + 262144);     // 256*128 f32
    float* ye              = (float*)(ws + 393216);
    float* xpart           = (float*)(ws + 524288);     // 256*512 f32
    float* ypart           = (float*)(ws + 1048576);
    unsigned short* W2h    = (unsigned short*)(ws + 1572864);  // 512*512 bf16
    unsigned short* w2hx   = (unsigned short*)(ws + 2097152);  // 128*320 bf16
    unsigned short* w2hy   = (unsigned short*)(ws + 2179072);

    // init accumulators + kl_loss output
    zero_kernel<<<128, 256, 0, stream>>>(pool_x, 32768);
    zero_kernel<<<128, 256, 0, stream>>>(pool_y, 32768);
    zero_kernel<<<1, 32, 0, stream>>>(out + 65536, 1);

    // weight precision conversion (f32 -> bf16 bits)
    cvt_bf16_kernel<<<(40960 + 255) / 256, 256, 0, stream>>>(cx_w2, w2hx, 40960);
    cvt_bf16_kernel<<<(40960 + 255) / 256, 256, 0, stream>>>(cy_w2, w2hy, 40960);
    cvt_bf16_kernel<<<(262144 + 255) / 256, 256, 0, stream>>>(h_w2, W2h, 262144);

    // fused conv towers (conv1 + conv2(WMMA) + pool-sum)
    tower_kernel<<<dim3(16, 256), 256, 0, stream>>>(x, cx_w1, cx_b1, w2hx, cx_b2, pool_x);
    tower_kernel<<<dim3(16, 256), 256, 0, stream>>>(y, cy_w1, cy_b1, w2hy, cy_b2, pool_y);

    // final linear of each tower
    fc_kernel<<<(32768 + 255) / 256, 256, 0, stream>>>(pool_x, cx_wf, cx_bf, xe);
    fc_kernel<<<(32768 + 255) / 256, 256, 0, stream>>>(pool_y, cy_wf, cy_bf, ye);

    // half-GEMMs against h_w1
    part_kernel<<<(131072 + 255) / 256, 256, 0, stream>>>(xe, h_w1, 0, xpart);
    part_kernel<<<(131072 + 255) / 256, 256, 0, stream>>>(ye, h_w1, 128, ypart);

    // fused pairwise MLP (dominant GEMM, WMMA)
    pair_kernel<<<1024, 128, 0, stream>>>(xpart, ypart, h_b1, W2h, h_b2, h_w3, h_b3, out);
}
